// BaseAttentionWrapper_91319594647581
// MI455X (gfx1250) — matrix operand order repaired and checked by reading, not compiled, not run on hardware
//
#include <hip/hip_runtime.h>

// ---------------------------------------------------------------------------
// Llama-style attention block for MI455X (gfx1250, wave32, WMMA).
// B=2, S=2048, D=4096, NH=32, NKV=8, HD=128, N_REP=4.
// All matmuls on v_wmma_f32_16x16x32_bf16; GEMM tiles staged into LDS with
// async global->LDS copies (ASYNCcnt) and double buffering.
// ---------------------------------------------------------------------------

typedef __attribute__((ext_vector_type(16))) __bf16 v16bf;
typedef __attribute__((ext_vector_type(8)))  __bf16 bf16x8;
typedef __attribute__((ext_vector_type(4)))  __bf16 bf16x4;
typedef __attribute__((ext_vector_type(2)))  __bf16 bf16x2;
typedef __attribute__((ext_vector_type(8)))  float  v8f;

static __device__ __forceinline__ v8f vzero8() {
  v8f z;
#pragma unroll
  for (int i = 0; i < 8; ++i) z[i] = 0.0f;
  return z;
}

// A-operand fragment (16x32 bf16): lanes 0-15 row M=lane, K = {0..7, 16..23};
// lanes 16-31 row M=lane-16, K = {8..15, 24..31}.
static __device__ __forceinline__ v16bf mkfrag(bf16x8 lo, bf16x8 hi) {
  v16bf r;
#pragma unroll
  for (int i = 0; i < 8; ++i) { r[i] = lo[i]; r[i + 8] = hi[i]; }
  return r;
}

static __device__ __forceinline__ v8f wmma_bf16(v16bf a, v16bf b, v8f c) {
  return __builtin_amdgcn_wmma_f32_16x16x32_bf16(
      /*neg_a=*/false, a, /*neg_b=*/false, b,
      /*c_mod=*/(short)0, c, /*reuse_a=*/false, /*reuse_b=*/false);
}

// Async 16-byte global -> LDS copy (gfx1250 GLOBAL_LOAD_ASYNC_TO_LDS_B128,
// GV addressing; tracked by ASYNCcnt). Per-lane addresses.
static __device__ __forceinline__ void async_cp16(const void* gptr, void* lptr) {
  unsigned la = (unsigned)(unsigned long long)lptr;       // wave LDS offset
  unsigned long long ga = (unsigned long long)gptr;
  asm volatile("global_load_async_to_lds_b128 %0, %1, off"
               :: "v"(la), "v"(ga) : "memory");
}

static __device__ __forceinline__ void wait_async0() {
  asm volatile("s_wait_asynccnt 0x0" ::: "memory");
}

// ---------------------------------------------------------------------------
// fp32 -> bf16 bulk convert (one pass; removes converts from GEMM hot loop,
// halves GEMM-side HBM bytes). n must be a multiple of 1024 (true here).
// ---------------------------------------------------------------------------
__global__ __launch_bounds__(256) void cvt_f32_bf16_kernel(
    const float* __restrict__ in, __bf16* __restrict__ out) {
  size_t i = ((size_t)blockIdx.x * 256 + threadIdx.x) * 4;
  float4 v = *(const float4*)(in + i);
  bf16x4 o = { (__bf16)v.x, (__bf16)v.y, (__bf16)v.z, (__bf16)v.w };
  *(bf16x4*)(out + i) = o;
}

// ---------------------------------------------------------------------------
// GEMM: C[M,N] = A[M,K] * W[N,K]^T   (bf16 in, f32 out)
// Block tile 128x128x32, 256 threads (8 waves, 2x4), wave tile 64x32.
// Double-buffered LDS; tiles staged with async global->LDS b128 copies.
// ---------------------------------------------------------------------------
__global__ __launch_bounds__(256) void gemm_bf16_async(
    const __bf16* __restrict__ A, const __bf16* __restrict__ W,
    float* __restrict__ C, int M, int N, int K) {
  __shared__ __bf16 sA[2][128][48];   // 32 cols + pad to 96B rows (16B align)
  __shared__ __bf16 sW[2][128][48];
  const int tid  = threadIdx.x;
  const int wave = tid >> 5;
  const int lane = tid & 31;
  const int half = lane >> 4;
  const int l16  = lane & 15;
  const int wm   = wave >> 2;          // 0..1 (64 rows)
  const int wn   = wave & 3;           // 0..3 (32 cols)
  const int bm   = blockIdx.y * 128;
  const int bn   = blockIdx.x * 128;

  // Per-thread staging slot: 128 rows x 4 chunks(16B) per tile, 2 per thread.
  auto stage = [&](int kt, int buf) {
#pragma unroll
    for (int i = 0; i < 2; ++i) {
      int idx = tid + (i << 8);        // 0..511
      int r   = idx >> 2;              // row 0..127
      int c   = (idx & 3) << 3;        // bf16 col 0,8,16,24 (16B chunks)
      async_cp16(A + (size_t)(bm + r) * K + kt + c, &sA[buf][r][c]);
      async_cp16(W + (size_t)(bn + r) * K + kt + c, &sW[buf][r][c]);
    }
  };

  v8f acc[4][2];
#pragma unroll
  for (int i = 0; i < 4; ++i)
#pragma unroll
    for (int j = 0; j < 2; ++j) acc[i][j] = vzero8();

  const int nkt = K >> 5;
  stage(0, 0);
  for (int t = 0; t < nkt; ++t) {
    const int cur = t & 1;
    wait_async0();                     // tile t landed in LDS (per-wave)
    __syncthreads();                   // visible to all waves; prev buf free
    if (t + 1 < nkt) stage((t + 1) << 5, cur ^ 1);   // prefetch overlaps WMMA

    v16bf af[4], bf[2];
#pragma unroll
    for (int i = 0; i < 4; ++i) {      // A: interleaved K layout
      int r = wm * 64 + i * 16 + l16;
      bf16x8 lo = *(const bf16x8*)(&sA[cur][r][half * 8]);
      bf16x8 hi = *(const bf16x8*)(&sA[cur][r][half * 8 + 16]);
      af[i] = mkfrag(lo, hi);
    }
#pragma unroll
    for (int j = 0; j < 2; ++j) {      // B: interleaved K layout
      int r = wn * 32 + j * 16 + l16;
      bf16x8 lo = *(const bf16x8*)(&sW[cur][r][half * 8]);
      bf16x8 hi = *(const bf16x8*)(&sW[cur][r][half * 8 + 16]);
      bf[j] = mkfrag(lo, hi);
    }
#pragma unroll
    for (int i = 0; i < 4; ++i)
#pragma unroll
      for (int j = 0; j < 2; ++j)
        acc[i][j] = wmma_bf16(af[i], bf[j], acc[i][j]);
  }

  // C/D layout: VGPR r -> M = r + 8*(lane>=16), lane%16 -> N.
#pragma unroll
  for (int i = 0; i < 4; ++i)
#pragma unroll
    for (int j = 0; j < 2; ++j)
#pragma unroll
      for (int r = 0; r < 8; ++r) {
        int row = bm + wm * 64 + i * 16 + r + 8 * half;
        int col = bn + wn * 32 + j * 16 + l16;
        C[(size_t)row * N + col] = acc[i][j][r];
      }
}

// ---------------------------------------------------------------------------
// RoPE on Q; write bf16 Q in [b, h, s, hd] layout. 2*2048*32*64 threads.
// ---------------------------------------------------------------------------
__global__ __launch_bounds__(256) void rope_q_kernel(
    const float* __restrict__ xq, const float* __restrict__ fr,
    __bf16* __restrict__ qb) {
  int idx = blockIdx.x * 256 + threadIdx.x;
  int d2 = idx & 63;
  int h  = (idx >> 6) & 31;
  int s  = (idx >> 11) & 2047;
  int b  = idx >> 22;
  size_t src = (size_t)(b * 2048 + s) * 4096 + h * 128 + 2 * d2;
  float x0 = xq[src], x1 = xq[src + 1];
  float c  = fr[(s * 64 + d2) * 2];
  float sn = fr[(s * 64 + d2) * 2 + 1];
  bf16x2 o = { (__bf16)(x0 * c - x1 * sn), (__bf16)(x1 * c + x0 * sn) };
  size_t dst = ((size_t)(b * 32 + h) * 2048 + s) * 128 + 2 * d2;
  *(bf16x2*)(qb + dst) = o;
}

// ---------------------------------------------------------------------------
// RoPE on K -> [b, hk, s, hd] bf16; V -> transposed [b, hk, hd, s] bf16.
// 2*2048*8*64 threads.
// ---------------------------------------------------------------------------
__global__ __launch_bounds__(256) void rope_kv_kernel(
    const float* __restrict__ xk, const float* __restrict__ xv,
    const float* __restrict__ fr, __bf16* __restrict__ kb,
    __bf16* __restrict__ vt) {
  int idx = blockIdx.x * 256 + threadIdx.x;
  int d2 = idx & 63;
  int hk = (idx >> 6) & 7;
  int s  = (idx >> 9) & 2047;
  int b  = idx >> 20;
  size_t src = (size_t)(b * 2048 + s) * 1024 + hk * 128 + 2 * d2;
  float k0 = xk[src], k1 = xk[src + 1];
  float c  = fr[(s * 64 + d2) * 2];
  float sn = fr[(s * 64 + d2) * 2 + 1];
  bf16x2 o = { (__bf16)(k0 * c - k1 * sn), (__bf16)(k1 * c + k0 * sn) };
  size_t kdst = ((size_t)(b * 8 + hk) * 2048 + s) * 128 + 2 * d2;
  *(bf16x2*)(kb + kdst) = o;
  float v0 = xv[src], v1 = xv[src + 1];
  size_t vdst = ((size_t)(b * 8 + hk) * 128 + 2 * d2) * 2048 + s;
  vt[vdst]        = (__bf16)v0;
  vt[vdst + 2048] = (__bf16)v1;
}

// ---------------------------------------------------------------------------
// Flash attention: block = 4 waves (128 thr), grid (S/64, NH, B).
// Wave owns 16 query rows; per 32-key block: 8 WMMA (Q*K^T), online softmax
// (16-lane shfl reductions), P restripe via per-wave LDS tile (s_wait_dscnt),
// 8 WMMA (P*V). Causal trip count is per-wave (no block barriers in loop).
// Output written directly as bf16 for the wo GEMM.
// ---------------------------------------------------------------------------
__global__ __launch_bounds__(128) void flash_attn_kernel(
    const __bf16* __restrict__ Q, const __bf16* __restrict__ Kc,
    const __bf16* __restrict__ Vt, __bf16* __restrict__ O) {
  __shared__ __bf16 sP[4][16][48];
  const int wave = threadIdx.x >> 5;
  const int lane = threadIdx.x & 31;
  const int half = lane >> 4;
  const int l16  = lane & 15;
  const int b  = blockIdx.z;
  const int h  = blockIdx.y;
  const int hk = h >> 2;               // GQA n_rep = 4
  const int q0 = blockIdx.x * 64 + wave * 16;
  const float scale = 0.08838834764831845f;   // 1/sqrt(128)

  const __bf16* qb = Q  + ((size_t)(b * 32 + h) * 2048 + q0) * 128;
  const __bf16* kb = Kc + (size_t)(b * 8 + hk) * 2048 * 128;
  const __bf16* vb = Vt + (size_t)(b * 8 + hk) * 128 * 2048;

  v16bf qf[4];
#pragma unroll
  for (int kc = 0; kc < 4; ++kc) {
    const __bf16* p = qb + (size_t)l16 * 128 + kc * 32 + half * 8;
    qf[kc] = mkfrag(*(const bf16x8*)p, *(const bf16x8*)(p + 16));
  }

  v8f o[8];
#pragma unroll
  for (int j = 0; j < 8; ++j) o[j] = vzero8();
  float mrow[8], lrow[8];
#pragma unroll
  for (int r = 0; r < 8; ++r) { mrow[r] = -1e30f; lrow[r] = 0.0f; }

  const int nkb = (q0 + 47) >> 5;      // causal: key blocks with k0 <= q0+15
  for (int kbi = 0; kbi < nkb; ++kbi) {
    const int k0 = kbi * 32;

    v8f sc[2];
    sc[0] = vzero8(); sc[1] = vzero8();
#pragma unroll
    for (int nh = 0; nh < 2; ++nh) {
#pragma unroll
      for (int kc = 0; kc < 4; ++kc) {
        const __bf16* p =
            kb + (size_t)(k0 + nh * 16 + l16) * 128 + kc * 32 + half * 8;
        v16bf kf = mkfrag(*(const bf16x8*)p, *(const bf16x8*)(p + 16));
        sc[nh] = wmma_bf16(qf[kc], kf, sc[nh]);
      }
    }

#pragma unroll
    for (int nh = 0; nh < 2; ++nh) {
      int key = k0 + nh * 16 + l16;
#pragma unroll
      for (int r = 0; r < 8; ++r) {
        int qrow = q0 + r + 8 * half;
        float sv = sc[nh][r] * scale;
        if (key > qrow) sv += -1e9f;   // reference's additive -1e9 mask
        sc[nh][r] = sv;
      }
    }

#pragma unroll
    for (int r = 0; r < 8; ++r) {
      float mx = fmaxf(sc[0][r], sc[1][r]);
#pragma unroll
      for (int m = 1; m < 16; m <<= 1) mx = fmaxf(mx, __shfl_xor(mx, m, 32));
      float mnew = fmaxf(mrow[r], mx);
      float corr = __expf(mrow[r] - mnew);
      float p0 = __expf(sc[0][r] - mnew);
      float p1 = __expf(sc[1][r] - mnew);
      float ls = p0 + p1;
#pragma unroll
      for (int m = 1; m < 16; m <<= 1) ls += __shfl_xor(ls, m, 32);
      lrow[r] = lrow[r] * corr + ls;
      mrow[r] = mnew;
#pragma unroll
      for (int j = 0; j < 8; ++j) o[j][r] *= corr;
      int mr = r + 8 * half;           // D-layout -> row-major LDS tile
      sP[wave][mr][l16]      = (__bf16)p0;
      sP[wave][mr][16 + l16] = (__bf16)p1;
    }
    asm volatile("s_wait_dscnt 0x0" ::: "memory");
    v16bf pf = mkfrag(*(const bf16x8*)(&sP[wave][l16][half * 8]),
                      *(const bf16x8*)(&sP[wave][l16][half * 8 + 16]));

#pragma unroll
    for (int j = 0; j < 8; ++j) {
      const __bf16* p = vb + (size_t)(j * 16 + l16) * 2048 + k0 + half * 8;
      v16bf vf = mkfrag(*(const bf16x8*)p, *(const bf16x8*)(p + 16));
      o[j] = wmma_bf16(pf, vf, o[j]);
    }
  }

#pragma unroll
  for (int r = 0; r < 8; ++r) {
    float inv = 1.0f / lrow[r];
    int row = q0 + r + 8 * half;
#pragma unroll
    for (int j = 0; j < 8; ++j) {
      O[(size_t)(b * 2048 + row) * 4096 + h * 128 + j * 16 + l16] =
          (__bf16)(o[j][r] * inv);
    }
  }
}

// ---------------------------------------------------------------------------
// Launch: bf16 pre-converts -> QKV GEMMs -> RoPE/layout -> flash attention
// (bf16 out) -> wo GEMM. Workspace regions aliased by lifetime.
// Inputs: x, start_pos(ignored,0), freqs_cis, mask(ignored; causal applied
// analytically), wq, wk, wv, wo.
// ---------------------------------------------------------------------------
extern "C" void kernel_launch(void* const* d_in, const int* in_sizes, int n_in,
                              void* d_out, int out_size, void* d_ws,
                              size_t ws_size, hipStream_t stream) {
  (void)in_sizes; (void)n_in; (void)out_size; (void)ws_size;
  const float* x  = (const float*)d_in[0];
  const float* fr = (const float*)d_in[2];
  const float* wq = (const float*)d_in[4];
  const float* wk = (const float*)d_in[5];
  const float* wv = (const float*)d_in[6];
  const float* wo = (const float*)d_in[7];
  float* out = (float*)d_out;

  const size_t MB = 1ull << 20;
  char* ws = (char*)d_ws;
  __bf16* xbf  = (__bf16*)(ws + 0);          // 32MB; reused as attn-out bf16
  __bf16* wqb  = (__bf16*)(ws + 32  * MB);   // 32MB; reused as wo bf16
  __bf16* wkb  = (__bf16*)(ws + 64  * MB);   // 8MB
  __bf16* wvb  = (__bf16*)(ws + 72  * MB);   // 8MB
  float*  xq   = (float*) (ws + 80  * MB);   // 64MB
  float*  xk   = (float*) (ws + 144 * MB);   // 16MB
  float*  xv   = (float*) (ws + 160 * MB);   // 16MB
  __bf16* qb   = (__bf16*)(ws + 176 * MB);   // 32MB
  __bf16* kbb  = (__bf16*)(ws + 208 * MB);   // 8MB
  __bf16* vt   = (__bf16*)(ws + 216 * MB);   // 8MB  (total 224MB)
  __bf16* aob  = xbf;                        // alias: x dead after QKV GEMMs
  __bf16* wob  = wqb;                        // alias: wq dead after Q GEMM

  dim3 blk(256);
  // One-shot fp32 -> bf16 (16.7M elems -> 16384 blocks; 4.2M -> 4096).
  cvt_f32_bf16_kernel<<<dim3(16384), blk, 0, stream>>>(x,  xbf);
  cvt_f32_bf16_kernel<<<dim3(16384), blk, 0, stream>>>(wq, wqb);
  cvt_f32_bf16_kernel<<<dim3(4096),  blk, 0, stream>>>(wk, wkb);
  cvt_f32_bf16_kernel<<<dim3(4096),  blk, 0, stream>>>(wv, wvb);

  // QKV projections: M = B*S = 4096 rows, K = 4096.
  gemm_bf16_async<<<dim3(32, 32), blk, 0, stream>>>(xbf, wqb, xq, 4096, 4096, 4096);
  gemm_bf16_async<<<dim3(8, 32),  blk, 0, stream>>>(xbf, wkb, xk, 4096, 1024, 4096);
  gemm_bf16_async<<<dim3(8, 32),  blk, 0, stream>>>(xbf, wvb, xv, 4096, 1024, 4096);

  rope_q_kernel<<<dim3(32768), blk, 0, stream>>>(xq, fr, qb);
  rope_kv_kernel<<<dim3(8192), blk, 0, stream>>>(xk, xv, fr, kbb, vt);

  // Attention writes bf16 directly into the (now dead) xbf region.
  flash_attn_kernel<<<dim3(32, 32, 2), dim3(128), 0, stream>>>(qb, kbb, vt, aob);

  // wo -> bf16 into the (now dead) wqb region, then output projection.
  cvt_f32_bf16_kernel<<<dim3(16384), blk, 0, stream>>>(wo, wob);
  gemm_bf16_async<<<dim3(32, 32), blk, 0, stream>>>(aob, wob, out, 4096, 4096, 4096);
}
